// IDynamicDWConv_19000935317707
// MI455X (gfx1250) — matrix operand order repaired and checked by reading, not compiled, not run on hardware
//
#include <hip/hip_runtime.h>

typedef __attribute__((ext_vector_type(16))) __bf16 v16bf;
typedef __attribute__((ext_vector_type(8)))  __bf16 v8bf;
typedef __attribute__((ext_vector_type(2)))  __bf16 v2bf;
typedef __attribute__((ext_vector_type(8)))  float  v8f;
typedef __attribute__((ext_vector_type(4)))  float  v4f;
typedef __attribute__((ext_vector_type(2)))  float  v2f;

#define B_   8
#define C_   256
#define H_   128
#define W_   128
#define HW_  (H_ * W_)      // 16384
#define CR_  64
#define G_   16
#define GC_  16
#define M2_  144            // K*K*G

#define LDA1 264            // ldsA row stride (bf16) kernel1: 528B, 16B-aligned, low-conflict
#define LDB1 40             // ldsB row stride (bf16) kernel1: 80B
#define LDA3 72             // kernel3 A stride: 144B
#define LDB3 72             // kernel3 B stride: 144B

union FragBF { v16bf v; v8bf h[2]; };

// ---------------------------------------------------------------------------
// Kernel 1: 1x1 conv (C=256 -> CR=64) as bf16 WMMA GEMM.
// grid.x = B * (HW/128) = 1024, block = 256 (8 waves).
// ---------------------------------------------------------------------------
__global__ __launch_bounds__(256)
void conv1_gemm(const float* __restrict__ x, const float* __restrict__ w1,
                const float* __restrict__ b1, float* __restrict__ y) {
  __shared__ __align__(16) __bf16 ldsA[CR_ * LDA1];   // [m][k]
  __shared__ __align__(16) __bf16 ldsB[128 * LDB1];   // [n][k]

  const int tid = threadIdx.x;
  const int blk = blockIdx.x;
  const int b   = blk >> 7;
  const int n0  = (blk & 127) * 128;

  // stage W1 fp32 -> bf16: 8192 (k,k+1) pairs, float2 loads + packed v2bf stores
  for (int i = 0; i < 32; ++i) {
    int p = tid + i * 256;
    int m = p >> 7;            // 128 pairs per row
    int j = p & 127;           // k = 2j
    v2f f = ((const v2f*)w1)[m * 128 + j];
    v2bf q; q[0] = (__bf16)f[0]; q[1] = (__bf16)f[1];
    *(v2bf*)&ldsA[m * LDA1 + 2 * j] = q;
  }

  const int lane = tid & 31;
  const int wv   = tid >> 5;
  const int half = lane >> 4;
  const int l15  = lane & 15;

  v8f acc[4] = {};
  const float* xb = x + (size_t)b * C_ * HW_;

  for (int k0 = 0; k0 < C_; k0 += 32) {
    __syncthreads();
    // stage x tile (32K x 128N): 512 units = (k-pair, n-quad); v4f loads, packed stores
    for (int i = 0; i < 2; ++i) {
      int u  = tid + i * 256;
      int kp = u >> 5;          // k = 2*kp
      int nq = u & 31;          // n = 4*nq
      const float* r0 = xb + (size_t)(k0 + 2 * kp) * HW_ + n0 + nq * 4;
      v4f lo = *(const v4f*)r0;
      v4f hi = *(const v4f*)(r0 + HW_);
#pragma unroll
      for (int j = 0; j < 4; ++j) {
        v2bf q; q[0] = (__bf16)lo[j]; q[1] = (__bf16)hi[j];
        *(v2bf*)&ldsB[(nq * 4 + j) * LDB1 + 2 * kp] = q;
      }
    }
    // prefetch next K tile while this one is consumed
    if (k0 + 32 < C_)
      __builtin_prefetch(xb + (size_t)(k0 + 32 + (tid & 31)) * HW_ + n0 + ((tid >> 5) << 4), 0, 0);
    __syncthreads();

    FragBF bf;
    {
      const __bf16* p = &ldsB[(wv * 16 + l15) * LDB1 + half * 16];
      bf.h[0] = *(const v8bf*)(p);
      bf.h[1] = *(const v8bf*)(p + 8);
    }
#pragma unroll
    for (int mt = 0; mt < 4; ++mt) {
      FragBF af;
      const __bf16* q = &ldsA[(mt * 16 + l15) * LDA1 + k0 + half * 16];
      af.h[0] = *(const v8bf*)(q);
      af.h[1] = *(const v8bf*)(q + 8);
      acc[mt] = __builtin_amdgcn_wmma_f32_16x16x32_bf16(
          false, af.v, false, bf.v, (short)0, acc[mt], false, false);
    }
  }

  const int nglob = n0 + wv * 16 + l15;
#pragma unroll
  for (int mt = 0; mt < 4; ++mt)
#pragma unroll
    for (int r = 0; r < 8; ++r) {
      int m = mt * 16 + half * 8 + r;
      y[((size_t)b * CR_ + m) * HW_ + nglob] = acc[mt][r] + b1[m];
    }
}

// ---------------------------------------------------------------------------
// Kernel 2: depthwise 3x3 + bias; each thread makes a horizontal output pair,
// packed 32-bit bf16 store. grid = B*CR*H*(W/2)/256 = 16384.
// ---------------------------------------------------------------------------
__global__ __launch_bounds__(256)
void dw3x3(const float* __restrict__ y, const float* __restrict__ wd,
           const float* __restrict__ bd, __bf16* __restrict__ z) {
  int p    = blockIdx.x * 256 + threadIdx.x;   // (b, c, h, w/2)
  int b    = p >> 19;                          // 64 * 8192
  int rem  = p & 524287;
  int c    = rem >> 13;                        // 8192 = H * W/2
  int hw2  = rem & 8191;
  int h    = hw2 >> 6;
  int w    = (hw2 & 63) * 2;

  const float* yc = y + ((size_t)b * CR_ + c) * HW_;
  float a0 = bd[c], a1 = a0;
#pragma unroll
  for (int kh = 0; kh < 3; ++kh) {
    int hh = h + kh - 1;
    if (hh < 0 || hh >= H_) continue;
    const float* row = yc + hh * W_;
#pragma unroll
    for (int kw = 0; kw < 3; ++kw) {
      float wt = wd[c * 9 + kh * 3 + kw];
      int w0 = w + kw - 1;
      if (w0 >= 0 && w0 < W_) a0 += wt * row[w0];
      int w1c = w + kw;                        // >= 0 always
      if (w1c < W_) a1 += wt * row[w1c];
    }
  }
  v2bf q; q[0] = (__bf16)a0; q[1] = (__bf16)a1;
  *(v2bf*)&z[((size_t)b * CR_ + c) * HW_ + h * W_ + w] = q;
}

// ---------------------------------------------------------------------------
// Kernel 3: 1x1 conv (64 -> 144) as bf16 WMMA GEMM; K=64 staged once.
// ---------------------------------------------------------------------------
__global__ __launch_bounds__(256)
void conv2_gemm(const __bf16* __restrict__ z, const float* __restrict__ w2,
                const float* __restrict__ b2, float* __restrict__ wg) {
  __shared__ __align__(16) __bf16 ldsA[M2_ * LDA3];  // 144 x 64
  __shared__ __align__(16) __bf16 ldsB[128 * LDB3];  // 128 x 64

  const int tid = threadIdx.x;
  const int blk = blockIdx.x;
  const int b   = blk >> 7;
  const int n0  = (blk & 127) * 128;

  // stage W2: 4608 pairs, float2 loads + packed stores
  for (int i = 0; i < 18; ++i) {
    int p = tid + i * 256;
    int m = p >> 5;            // 32 pairs per row
    int j = p & 31;
    v2f f = ((const v2f*)w2)[m * 32 + j];
    v2bf q; q[0] = (__bf16)f[0]; q[1] = (__bf16)f[1];
    *(v2bf*)&ldsA[m * LDA3 + 2 * j] = q;
  }
  // stage z tile (64K x 128N): 512 units = (k-pair, n-octet); v8bf loads
  const __bf16* zb = z + (size_t)b * CR_ * HW_;
  for (int i = 0; i < 2; ++i) {
    int u  = tid + i * 256;
    int kp = u >> 4;           // k = 2*kp, 0..31
    int no = (u & 15) * 8;
    const __bf16* r0 = zb + (size_t)(2 * kp) * HW_ + n0 + no;
    v8bf lo = *(const v8bf*)r0;
    v8bf hi = *(const v8bf*)(r0 + HW_);
#pragma unroll
    for (int j = 0; j < 8; ++j) {
      v2bf q; q[0] = lo[j]; q[1] = hi[j];
      *(v2bf*)&ldsB[(no + j) * LDB3 + 2 * kp] = q;
    }
  }
  __syncthreads();

  const int lane = tid & 31, wv = tid >> 5, half = lane >> 4, l15 = lane & 15;
  v8f acc[9] = {};
#pragma unroll
  for (int k0 = 0; k0 < 64; k0 += 32) {
    FragBF bf;
    const __bf16* p = &ldsB[(wv * 16 + l15) * LDB3 + k0 + half * 16];
    bf.h[0] = *(const v8bf*)(p);
    bf.h[1] = *(const v8bf*)(p + 8);
#pragma unroll
    for (int mt = 0; mt < 9; ++mt) {
      FragBF af;
      const __bf16* q = &ldsA[(mt * 16 + l15) * LDA3 + k0 + half * 16];
      af.h[0] = *(const v8bf*)(q);
      af.h[1] = *(const v8bf*)(q + 8);
      acc[mt] = __builtin_amdgcn_wmma_f32_16x16x32_bf16(
          false, af.v, false, bf.v, (short)0, acc[mt], false, false);
    }
  }
  const int nglob = n0 + wv * 16 + l15;
#pragma unroll
  for (int mt = 0; mt < 9; ++mt)
#pragma unroll
    for (int r = 0; r < 8; ++r) {
      int m = mt * 16 + half * 8 + r;
      wg[((size_t)b * M2_ + m) * HW_ + nglob] = acc[mt][r] + b2[m];
    }
}

// ---------------------------------------------------------------------------
// Kernel 4: involution. One block per (b, g, h). Weight row (9x128 fp32)
// staged global->LDS with CDNA5 async-to-LDS DMA (ASYNCcnt), x taps from L2.
// NOTE: wrow is the ONLY shared allocation here, so its LDS offset is 0.
// ---------------------------------------------------------------------------
__global__ __launch_bounds__(256)
void involution(const float* __restrict__ x, const float* __restrict__ wg,
                float* __restrict__ out) {
  __shared__ __align__(16) float wrow[9 * 128];
  const int tid = threadIdx.x;
  const int blk = blockIdx.x;               // B*G*H = 16384
  const int b   = blk / (G_ * H_);
  const int rem = blk % (G_ * H_);
  const int g   = rem / H_;
  const int h   = rem % H_;

  const float* wsrc = wg + ((size_t)b * M2_ + g * 9) * HW_ + h * W_;
  {
    unsigned long long base = (unsigned long long)wsrc;
#pragma unroll
    for (int i = 0; i < 2; ++i) {
      int e = tid + i * 256;                // 288 b128 transfers (9*128 floats)
      if (e < 288) {
        int t   = e >> 5;                   // tap index 0..8
        int col = (e & 31) * 4;
        unsigned goff = (unsigned)((t * (unsigned)HW_ + col) * 4u);
        unsigned loff = (unsigned)e * 16u;  // LDS byte offset of wrow[e*4]
        asm volatile("global_load_async_to_lds_b128 %0, %1, %2"
                     :: "v"(loff), "v"(goff), "s"(base) : "memory");
      }
    }
    asm volatile("s_wait_asynccnt 0x0" ::: "memory");
  }
  __syncthreads();

#pragma unroll
  for (int i = 0; i < 8; ++i) {
    int lin = tid + i * 256;                // 16 gc * 128 w
    int gc  = lin >> 7;
    int w   = lin & 127;
    const float* xc = x + ((size_t)b * C_ + g * GC_ + gc) * HW_;
    float acc = 0.f;
#pragma unroll
    for (int kh = 0; kh < 3; ++kh) {
      int hh = h + kh - 1;
      if (hh < 0 || hh >= H_) continue;
#pragma unroll
      for (int kw = 0; kw < 3; ++kw) {
        int ww = w + kw - 1;
        if (ww < 0 || ww >= W_) continue;
        acc += xc[hh * W_ + ww] * wrow[(kh * 3 + kw) * 128 + w];
      }
    }
    out[((size_t)b * C_ + g * GC_ + gc) * HW_ + h * W_ + w] = acc;
  }
}

// ---------------------------------------------------------------------------
extern "C" void kernel_launch(void* const* d_in, const int* in_sizes, int n_in,
                              void* d_out, int out_size, void* d_ws, size_t ws_size,
                              hipStream_t stream) {
  (void)in_sizes; (void)n_in; (void)out_size; (void)ws_size;

  const float* x  = (const float*)d_in[0];
  const float* w1 = (const float*)d_in[1];
  const float* b1 = (const float*)d_in[2];
  const float* wd = (const float*)d_in[3];
  const float* bd = (const float*)d_in[4];
  const float* w2 = (const float*)d_in[5];
  const float* b2 = (const float*)d_in[6];
  float* out = (float*)d_out;

  const size_t Y_BYTES = (size_t)B_ * CR_ * HW_ * sizeof(float);   // 32 MiB
  const size_t Z_BYTES = (size_t)B_ * CR_ * HW_ * sizeof(__bf16);  // 16 MiB
  char* ws = (char*)d_ws;
  float*  y     = (float*)ws;
  __bf16* z     = (__bf16*)(ws + Y_BYTES);
  float*  wgbuf = (float*)(ws + Y_BYTES + Z_BYTES);                // 72 MiB

  conv1_gemm<<<dim3(B_ * (HW_ / 128)), dim3(256), 0, stream>>>(x, w1, b1, y);
  dw3x3<<<dim3((B_ * CR_ * H_ * (W_ / 2)) / 256), dim3(256), 0, stream>>>(y, wd, bd, z);
  conv2_gemm<<<dim3(B_ * (HW_ / 128)), dim3(256), 0, stream>>>(z, w2, b2, wgbuf);
  involution<<<dim3(B_ * G_ * H_), dim3(256), 0, stream>>>(x, wgbuf, out);
}
